// snowball_62878321213489
// MI455X (gfx1250) — matrix operand order repaired and checked by reading, not compiled, not run on hardware
//
#include <hip/hip_runtime.h>

typedef __attribute__((ext_vector_type(8)))  __bf16 v8bf;
typedef __attribute__((ext_vector_type(16))) __bf16 v16bf;
typedef __attribute__((ext_vector_type(8)))  float  v8f;

__device__ __forceinline__ unsigned short f2bf_rne(float f) {
  unsigned int u = __float_as_uint(f);
  u += 0x7FFFu + ((u >> 16) & 1u);   // round-to-nearest-even
  return (unsigned short)(u >> 16);
}

__global__ void fill_zero(float* __restrict__ p, long n) {
  long t = (long)blockIdx.x * blockDim.x + threadIdx.x;
  if (t < n) p[t] = 0.f;
}

// copy fp32 [n,d] rows into bf16 cat columns [col0, col0+d)
__global__ void cast_rows_to_cat(const float* __restrict__ X, unsigned short* __restrict__ cat,
                                 int n, int d, int col0, int ldc) {
  long t = (long)blockIdx.x * blockDim.x + threadIdx.x;
  if (t >= (long)n * d) return;
  int r = (int)(t / d), c = (int)(t % d);
  cat[(long)r * ldc + col0 + c] = f2bf_rne(X[t]);
}

// h = relu(H + b) -> bf16 cat columns [col0, col0+d)
__global__ void bias_relu_to_cat(const float* __restrict__ H, const float* __restrict__ b,
                                 unsigned short* __restrict__ cat, int n, int d, int col0, int ldc) {
  long t = (long)blockIdx.x * blockDim.x + threadIdx.x;
  if (t >= (long)n * d) return;
  int r = (int)(t / d), c = (int)(t % d);
  float v = H[t] + b[c];
  v = fmaxf(v, 0.f);
  cat[(long)r * ldc + col0 + c] = f2bf_rne(v);
}

// Wt[o, i] = bf16(W[i, o]) for o < dout, zero-pad rows dout..dpad
__global__ void transpose_cast_w(const float* __restrict__ W, unsigned short* __restrict__ Wt,
                                 int din, int dout, int dpad) {
  long t = (long)blockIdx.x * blockDim.x + threadIdx.x;
  if (t >= (long)dpad * din) return;
  int o = (int)(t / din), i = (int)(t % din);
  float v = (o < dout) ? W[(long)i * dout + o] : 0.f;
  Wt[t] = f2bf_rne(v);
}

// C[n, dout] = A_bf16[n, 0:din (lda stride)] @ Bt_bf16[dout, din]^T, fp32 accum.
// One wave per 16x16 output tile; K-loop step 32 with v_wmma_f32_16x16x32_bf16.
__global__ void gemm_bf16_wmma(const unsigned short* __restrict__ A,
                               const unsigned short* __restrict__ Bt,
                               float* __restrict__ C,
                               int n, int din, int dout, int lda) {
  const int gw   = (int)(((long)blockIdx.x * blockDim.x + threadIdx.x) >> 5);
  const int lane = threadIdx.x & 31;
  const int tiles_n = dout >> 4;
  const int tiles_m = n >> 4;
  if (gw >= tiles_m * tiles_n) return;            // wave-uniform: EXEC stays all-1 for WMMA
  const int tm = gw / tiles_n;
  const int tn = gw % tiles_n;
  const int half = lane >> 4;                     // which 16-lane half
  const int l16  = lane & 15;

  // A operand, in this code's contiguous k order: lanes 0-15 hold M=l16, K 0-15;
  //                            lanes 16-31 hold K 16-31.
  const unsigned short* arow = A  + (long)(tm * 16 + l16) * lda + half * 16;
  // B operand, same contiguous k order as A: lanes 0-15 hold N=l16, K 0-15; lanes 16-31 K 16-31.
  const unsigned short* bcol = Bt + (long)(tn * 16 + l16) * din + half * 16;

  v8f acc = {};
  for (int k0 = 0; k0 < din; k0 += 32) {
    v8bf  a0 = *(const v8bf*)(arow + k0);
    v8bf  a1 = *(const v8bf*)(arow + k0 + 8);
    v16bf a  = __builtin_shufflevector(a0, a1, 0,1,2,3,4,5,6,7,8,9,10,11,12,13,14,15);
    v16bf b  = *(const v16bf*)(bcol + k0);
    if (k0 + 32 < din) {
      __builtin_prefetch(arow + k0 + 32, 0, 0);   // global_prefetch_b8
      __builtin_prefetch(bcol + k0 + 32, 0, 0);
    }
    acc = __builtin_amdgcn_wmma_f32_16x16x32_bf16(
        /*neg_a=*/false, a, /*neg_b=*/false, b,
        /*c_mod=*/(short)0, acc, /*reuse_a=*/false, /*reuse_b=*/false);
  }

  // D layout: VGPR r -> row M = half*8 + r, col N = l16
  float* cp = C + (long)(tm * 16 + half * 8) * dout + tn * 16 + l16;
#pragma unroll
  for (int r = 0; r < 8; ++r) cp[(long)r * dout] = acc[r];
}

// Y[dst[e], :] += val[e] * X[src[e], :], float4 per thread, atomic scatter.
__global__ void spmm4(const int* __restrict__ src, const int* __restrict__ dst,
                      const float* __restrict__ val, const float* __restrict__ X,
                      float* __restrict__ Y, int E, int d, int ldx) {
  const int per = d >> 2;
  long t = (long)blockIdx.x * blockDim.x + threadIdx.x;
  if (t >= (long)E * per) return;
  int e  = (int)(t / per);
  int c4 = ((int)(t % per)) << 2;
  int s = src[e], q = dst[e];
  float v = val[e];
  float4 xv = *(const float4*)(X + (long)s * ldx + c4);
  float* y = Y + (long)q * d + c4;
  atomicAdd(y + 0, v * xv.x);
  atomicAdd(y + 1, v * xv.y);
  atomicAdd(y + 2, v * xv.z);
  atomicAdd(y + 3, v * xv.w);
}

__global__ void add_bias(float* __restrict__ Y, const float* __restrict__ b, int n, int d) {
  long t = (long)blockIdx.x * blockDim.x + threadIdx.x;
  if (t >= (long)n * d) return;
  Y[t] += b[(int)(t % d)];
}

static inline int cdiv(long a, long b) { return (int)((a + b - 1) / b); }

extern "C" void kernel_launch(void* const* d_in, const int* in_sizes, int n_in,
                              void* d_out, int out_size, void* d_ws, size_t ws_size,
                              hipStream_t stream) {
  const float* x    = (const float*)d_in[0];
  const int*   esrc = (const int*)d_in[1];
  const int*   edst = (const int*)d_in[2];
  const float* eval = (const float*)d_in[3];
  const float* W[5]    = {(const float*)d_in[4],  (const float*)d_in[6],
                          (const float*)d_in[8],  (const float*)d_in[10],
                          (const float*)d_in[12]};
  const float* bias[5] = {(const float*)d_in[5],  (const float*)d_in[7],
                          (const float*)d_in[9],  (const float*)d_in[11],
                          (const float*)d_in[13]};

  const int N = 50000, E = 800000, F = 512, H = 128, L = 4, C = 40;
  const int CAT = F + L * H;   // 1024
  const int CP  = 48;          // Wout padded 40 -> 48 (3 WMMA column tiles)

  char* ws = (char*)d_ws;
  unsigned short* cat = (unsigned short*)ws;  ws += (size_t)N * CAT * sizeof(unsigned short);
  float* gemm_out     = (float*)ws;           ws += (size_t)N * H * sizeof(float);
  float* spmm_out     = (float*)ws;           ws += (size_t)N * H * sizeof(float);
  unsigned short* Wt  = (unsigned short*)ws;  ws += (size_t)CAT * H * sizeof(unsigned short);

  const int BT = 256;   // 8 wave32s per block

  // x -> cat[:, 0:512)
  cast_rows_to_cat<<<cdiv((long)N * F, BT), BT, 0, stream>>>(x, cat, N, F, 0, CAT);

  for (int k = 0; k < L; ++k) {
    const int din = F + k * H;
    transpose_cast_w<<<cdiv((long)H * din, BT), BT, 0, stream>>>(W[k], Wt, din, H, H);
    const long waves = (long)(N / 16) * (H / 16);
    gemm_bf16_wmma<<<cdiv(waves * 32, BT), BT, 0, stream>>>(cat, Wt, gemm_out, N, din, H, CAT);
    fill_zero<<<cdiv((long)N * H, BT), BT, 0, stream>>>(spmm_out, (long)N * H);
    spmm4<<<cdiv((long)E * (H / 4), BT), BT, 0, stream>>>(esrc, edst, eval, gemm_out, spmm_out, E, H, H);
    bias_relu_to_cat<<<cdiv((long)N * H, BT), BT, 0, stream>>>(spmm_out, bias[k], cat, N, H, F + k * H, CAT);
  }

  // Output layer
  transpose_cast_w<<<cdiv((long)CP * CAT, BT), BT, 0, stream>>>(W[4], Wt, CAT, C, CP);
  const long waves = (long)(N / 16) * (CP / 16);
  gemm_bf16_wmma<<<cdiv(waves * 32, BT), BT, 0, stream>>>(cat, Wt, gemm_out, N, CAT, CP, CAT);
  float* out = (float*)d_out;
  fill_zero<<<cdiv((long)N * C, BT), BT, 0, stream>>>(out, (long)N * C);
  spmm4<<<cdiv((long)E * (C / 4), BT), BT, 0, stream>>>(esrc, edst, eval, gemm_out, out, E, C, CP);
  add_bias<<<cdiv((long)N * C, BT), BT, 0, stream>>>(out, bias[4], N, C);
}
